// TarMAC_Comm_60662118089448
// MI455X (gfx1250) — matrix operand order repaired and checked by reading, not compiled, not run on hardware
//
#include <hip/hip_runtime.h>

typedef __attribute__((ext_vector_type(16))) __bf16 v16bf;
typedef __attribute__((ext_vector_type(8)))  __bf16 v8bf;
typedef __attribute__((ext_vector_type(8)))  float  v8f;
typedef __attribute__((ext_vector_type(4)))  int    v4i;

#define D_MODEL 512
#define N_AGENTS_C 8192

#if defined(__AMDGCN__) && __has_builtin(__builtin_amdgcn_global_load_async_to_lds_b128) && \
    __has_builtin(__builtin_amdgcn_s_wait_asynccnt)
#define USE_ASYNC_LDS 1
#else
#define USE_ASYNC_LDS 0
#endif

__device__ __forceinline__ unsigned short f2bf(float f) {
  union { float f; unsigned int u; } c; c.f = f;
  unsigned int u = c.u;
  unsigned int r = (u + 0x7FFFu + ((u >> 16) & 1u)) >> 16;
  return (unsigned short)r;
}

__device__ __forceinline__ v16bf mkfrag(v8bf lo, v8bf hi) {
  v16bf r;
#pragma unroll
  for (int i = 0; i < 8; ++i) { r[i] = lo[i]; r[8 + i] = hi[i]; }
  return r;
}

// Copy 32 bytes global -> LDS. Async (ASYNCcnt-tracked, direct-to-LDS) when
// available, else synchronous vector copy through VGPRs.
__device__ __forceinline__ void g2l_32B(unsigned short* l, const unsigned short* g) {
#if USE_ASYNC_LDS
  __builtin_amdgcn_global_load_async_to_lds_b128(
      (__attribute__((address_space(1))) v4i*)g,
      (__attribute__((address_space(3))) v4i*)l, 0, 0);
  __builtin_amdgcn_global_load_async_to_lds_b128(
      (__attribute__((address_space(1))) v4i*)(g + 8),
      (__attribute__((address_space(3))) v4i*)(l + 8), 0, 0);
#else
  *(v16bf*)l = *(const v16bf*)g;
#endif
}

__device__ __forceinline__ void stage_fence() {
#if USE_ASYNC_LDS
  __builtin_amdgcn_s_wait_asynccnt(0);
#endif
}

// ---------------------------------------------------------------------------
// Generic GEMM: C[M,N] = act(A[M,K] @ W[N,K]^T + bias[N]), bf16 in/out, f32 acc
// Block: 256 thr (8 waves), tile 128x128, BK=32. Wave = 32x64 (2x4 wmma tiles).
// Double-buffered LDS staging via async-to-LDS; one barrier per K-step.
// ---------------------------------------------------------------------------
__global__ __launch_bounds__(256) void gemm_bias_act(
    const unsigned short* __restrict__ A, const unsigned short* __restrict__ W,
    const float* __restrict__ bias, unsigned short* __restrict__ C,
    int M, int N, int K, int act)
{
  __shared__ __attribute__((aligned(32))) unsigned short As[2 * 128 * 32];
  __shared__ __attribute__((aligned(32))) unsigned short Ws[2 * 128 * 32];

  const int tid  = threadIdx.x;
  const int wvid = tid >> 5;
  const int lane = tid & 31;
  const int n    = lane & 15;   // wmma column within 16x16 tile
  const int kg   = lane >> 4;   // lane half-group
  const int wrow = wvid & 3;    // 4 row-groups of 32 rows
  const int wcol = wvid >> 2;   // 2 col-groups of 64 cols
  const int mBase = blockIdx.y * 128;
  const int nBase = blockIdx.x * 128;

  v8f acc[2][4] = {};

  const int srow  = tid >> 1;   // staging: 128 rows x 2 chunks of 16 halves
  const int shalf = tid & 1;

  // prefetch tile 0 into buffer 0
  g2l_32B(As + srow * 32 + shalf * 16,
          A + (size_t)(mBase + srow) * K + shalf * 16);
  g2l_32B(Ws + srow * 32 + shalf * 16,
          W + (size_t)(nBase + srow) * K + shalf * 16);

  for (int k0 = 0; k0 < K; k0 += 32) {
    const int cur = (k0 >> 5) & 1;
    stage_fence();
    __syncthreads();
    if (k0 + 32 < K) {
      const int nxt = cur ^ 1;
      g2l_32B(As + nxt * 4096 + srow * 32 + shalf * 16,
              A + (size_t)(mBase + srow) * K + (k0 + 32) + shalf * 16);
      g2l_32B(Ws + nxt * 4096 + srow * 32 + shalf * 16,
              W + (size_t)(nBase + srow) * K + (k0 + 32) + shalf * 16);
    }

    const unsigned short* Ab = As + cur * 4096;
    const unsigned short* Wb = Ws + cur * 4096;
    v16bf af[2], bfm[4];
#pragma unroll
    for (int mt = 0; mt < 2; ++mt) {
      const unsigned short* p = Ab + (wrow * 32 + mt * 16 + n) * 32 + kg * 16;
      af[mt] = mkfrag(*(const v8bf*)p, *(const v8bf*)(p + 8));
    }
#pragma unroll
    for (int nt = 0; nt < 4; ++nt)
      bfm[nt] = *(const v16bf*)(Wb + (wcol * 64 + nt * 16 + n) * 32 + kg * 16);
#pragma unroll
    for (int mt = 0; mt < 2; ++mt)
#pragma unroll
      for (int nt = 0; nt < 4; ++nt)
        acc[mt][nt] = __builtin_amdgcn_wmma_f32_16x16x32_bf16(
            false, af[mt], false, bfm[nt], (short)0, acc[mt][nt], false, false);
  }

#pragma unroll
  for (int nt = 0; nt < 4; ++nt) {
    const int col = nBase + wcol * 64 + nt * 16 + n;
    const float bv = bias[col];
#pragma unroll
    for (int mt = 0; mt < 2; ++mt) {
#pragma unroll
      for (int r = 0; r < 8; ++r) {
        const int row = mBase + wrow * 32 + mt * 16 + kg * 8 + r;
        float v = acc[mt][nt][r] + bv;
        if (act) v = tanhf(v);
        C[(size_t)row * N + col] = f2bf(v);
      }
    }
  }
}

// ---------------------------------------------------------------------------
// Fused flash attention: Out = softmax(Q K^T / sqrt(D)) V
// 4 waves/block, each wave owns 16 query rows and full D=512 output (32 v8f
// accumulators). Q resident in registers in A-fragment layout. K and V^T
// staged via double-buffered async-to-LDS 64x64 bf16 tiles; P through LDS.
// ---------------------------------------------------------------------------
__global__ __launch_bounds__(128) void flash_attn(
    const unsigned short* __restrict__ Q, const unsigned short* __restrict__ Kb,
    const unsigned short* __restrict__ Vt, float* __restrict__ Out, int N)
{
  __shared__ __attribute__((aligned(32))) unsigned short Ks[2 * 64 * 64];
  __shared__ __attribute__((aligned(32))) unsigned short Vs[2 * 64 * 64];
  __shared__ __attribute__((aligned(32))) unsigned short Ps[4 * 16 * 64];

  const int tid  = threadIdx.x;
  const int wv   = tid >> 5;
  const int lane = tid & 31;
  const int n    = lane & 15;
  const int kg   = lane >> 4;
  const int q0   = blockIdx.x * 64 + wv * 16;

  // Q fragments: 16 k-steps of 32 over D=512
  v16bf qf[16];
  {
    const unsigned short* qrow = Q + (size_t)(q0 + n) * D_MODEL;
#pragma unroll
    for (int ks = 0; ks < 16; ++ks) {
      const unsigned short* p = qrow + ks * 32 + kg * 16;
      qf[ks] = mkfrag(*(const v8bf*)p, *(const v8bf*)(p + 8));
    }
  }

  v8f o[32] = {};
  float rmax[8], rsum[8];
#pragma unroll
  for (int r = 0; r < 8; ++r) { rmax[r] = -3.0e38f; rsum[r] = 0.f; }

  const int srow  = tid >> 1;   // staging: 64 rows x 2 chunks of 32 halves
  const int shalf = tid & 1;
  const float sc = 0.04419417382415922f;  // 1/sqrt(512)

  for (int kb = 0; kb < N; kb += 64) {
    // ---- S = Q K^T tile (16 x 64 per wave), double-buffered K chunks ----
    v8f s[4] = {};
    {
      const unsigned short* src = Kb + (size_t)(kb + srow) * D_MODEL + shalf * 32;
      g2l_32B(Ks + srow * 64 + shalf * 32, src);
      g2l_32B(Ks + srow * 64 + shalf * 32 + 16, src + 16);
    }
#pragma unroll
    for (int dk = 0; dk < D_MODEL; dk += 64) {
      const int cur = (dk >> 6) & 1;
      stage_fence();
      __syncthreads();
      if (dk + 64 < D_MODEL) {
        const unsigned short* src =
            Kb + (size_t)(kb + srow) * D_MODEL + (dk + 64) + shalf * 32;
        g2l_32B(Ks + (cur ^ 1) * 4096 + srow * 64 + shalf * 32, src);
        g2l_32B(Ks + (cur ^ 1) * 4096 + srow * 64 + shalf * 32 + 16, src + 16);
      }
      const unsigned short* Kbuf = Ks + cur * 4096;
#pragma unroll
      for (int ks2 = 0; ks2 < 2; ++ks2) {
        v16bf af = qf[(dk >> 5) + ks2];
#pragma unroll
        for (int nt = 0; nt < 4; ++nt) {
          v16bf bfm = *(const v16bf*)(Kbuf + (nt * 16 + n) * 64 + ks2 * 32 + kg * 16);
          s[nt] = __builtin_amdgcn_wmma_f32_16x16x32_bf16(
              false, af, false, bfm, (short)0, s[nt], false, false);
        }
      }
    }

    // ---- online softmax; write P (bf16) to LDS in A-fragment-friendly layout ----
#pragma unroll
    for (int r = 0; r < 8; ++r) {
      float mb = -3.0e38f;
#pragma unroll
      for (int nt = 0; nt < 4; ++nt) mb = fmaxf(mb, s[nt][r]);
#pragma unroll
      for (int off = 1; off < 16; off <<= 1) mb = fmaxf(mb, __shfl_xor(mb, off, 16));
      mb *= sc;
      const float mnew = fmaxf(rmax[r], mb);
      const float corr = __expf(rmax[r] - mnew);
      rmax[r] = mnew;
      float psum = 0.f;
#pragma unroll
      for (int nt = 0; nt < 4; ++nt) {
        const float p = __expf(s[nt][r] * sc - mnew);
        psum += p;
        Ps[(wv * 16 + kg * 8 + r) * 64 + nt * 16 + n] = f2bf(p);
      }
#pragma unroll
      for (int off = 1; off < 16; off <<= 1) psum += __shfl_xor(psum, off, 16);
      rsum[r] = rsum[r] * corr + psum;
#pragma unroll
      for (int t = 0; t < 32; ++t) o[t][r] *= corr;
    }

    // ---- O += P V using V^T tiles (contract over 64 keys), double-buffered ----
    {
      const unsigned short* src = Vt + (size_t)srow * N + kb + shalf * 32;
      g2l_32B(Vs + srow * 64 + shalf * 32, src);
      g2l_32B(Vs + srow * 64 + shalf * 32 + 16, src + 16);
    }
#pragma unroll
    for (int dv = 0; dv < D_MODEL; dv += 64) {
      const int cur = (dv >> 6) & 1;
      stage_fence();
      __syncthreads();
      if (dv + 64 < D_MODEL) {
        const unsigned short* src =
            Vt + (size_t)(dv + 64 + srow) * N + kb + shalf * 32;
        g2l_32B(Vs + (cur ^ 1) * 4096 + srow * 64 + shalf * 32, src);
        g2l_32B(Vs + (cur ^ 1) * 4096 + srow * 64 + shalf * 32 + 16, src + 16);
      }
      const unsigned short* Vbuf = Vs + cur * 4096;
#pragma unroll
      for (int ks2 = 0; ks2 < 2; ++ks2) {
        const unsigned short* pr = Ps + (wv * 16 + n) * 64 + ks2 * 32 + kg * 16;
        v16bf af = mkfrag(*(const v8bf*)pr, *(const v8bf*)(pr + 8));
#pragma unroll
        for (int nt = 0; nt < 4; ++nt) {
          v16bf bfm = *(const v16bf*)(Vbuf + (nt * 16 + n) * 64 + ks2 * 32 + kg * 16);
          o[(dv >> 4) + nt] = __builtin_amdgcn_wmma_f32_16x16x32_bf16(
              false, af, false, bfm, (short)0, o[(dv >> 4) + nt], false, false);
        }
      }
    }
    stage_fence();
    __syncthreads();
  }

  // ---- epilogue: divide by row sums, write f32 ----
#pragma unroll
  for (int r = 0; r < 8; ++r) {
    const float inv = 1.f / rsum[r];
    const size_t row = (size_t)(q0 + kg * 8 + r);
#pragma unroll
    for (int t = 0; t < 32; ++t)
      Out[row * D_MODEL + t * 16 + n] = o[t][r] * inv;
  }
}

// ---------------------------------------------------------------------------
// Helpers
// ---------------------------------------------------------------------------
__global__ void cvt_f32_bf16(const float* __restrict__ in,
                             unsigned short* __restrict__ out, int nElem) {
  int i = blockIdx.x * blockDim.x + threadIdx.x;
  if (i < nElem) out[i] = f2bf(in[i]);
}

__global__ void transpose_bf16(const unsigned short* __restrict__ in,
                               unsigned short* __restrict__ out, int rows, int cols) {
  __shared__ unsigned short tile[32][33];
  const int c0 = blockIdx.x * 32, r0 = blockIdx.y * 32;
  const int tx = threadIdx.x, ty = threadIdx.y;
#pragma unroll
  for (int i = 0; i < 32; i += 8)
    tile[ty + i][tx] = in[(size_t)(r0 + ty + i) * cols + c0 + tx];
  __syncthreads();
#pragma unroll
  for (int i = 0; i < 32; i += 8)
    out[(size_t)(c0 + ty + i) * rows + r0 + tx] = tile[tx][ty + i];
}

__global__ void concat_comm_h(const float* __restrict__ comm,
                              const unsigned short* __restrict__ h,
                              unsigned short* __restrict__ cat, int total) {
  int i = blockIdx.x * blockDim.x + threadIdx.x;
  if (i >= total) return;
  int row = i >> 10, col = i & 1023;
  cat[i] = (col < 512) ? f2bf(comm[row * 512 + col]) : h[row * 512 + (col - 512)];
}

// ---------------------------------------------------------------------------
extern "C" void kernel_launch(void* const* d_in, const int* in_sizes, int n_in,
                              void* d_out, int out_size, void* d_ws, size_t ws_size,
                              hipStream_t stream) {
  (void)in_sizes; (void)n_in; (void)out_size; (void)ws_size;
  const float* hidden = (const float*)d_in[0];
  const float* f_kw1  = (const float*)d_in[1];
  const float* key_b1 = (const float*)d_in[2];
  const float* f_kw2  = (const float*)d_in[3];
  const float* key_b2 = (const float*)d_in[4];
  const float* f_vw1  = (const float*)d_in[5];
  const float* val_b1 = (const float*)d_in[6];
  const float* f_vw2  = (const float*)d_in[7];
  const float* val_b2 = (const float*)d_in[8];
  const float* f_qw1  = (const float*)d_in[9];
  const float* qry_b1 = (const float*)d_in[10];
  const float* f_qw2  = (const float*)d_in[11];
  const float* qry_b2 = (const float*)d_in[12];
  const float* f_mw1  = (const float*)d_in[13];
  const float* msg_b1 = (const float*)d_in[14];
  const float* f_mw2  = (const float*)d_in[15];
  const float* msg_b2 = (const float*)d_in[16];

  char* ws = (char*)d_ws;
  size_t off = 0;
  auto take = [&](size_t bytes) {
    char* p = ws + off;
    off += (bytes + 255) & ~(size_t)255;
    return p;
  };

  const int N = N_AGENTS_C, D = D_MODEL;
  unsigned short* kw1 = (unsigned short*)take((size_t)D * D * 2);
  unsigned short* kw2 = (unsigned short*)take((size_t)D * D * 2);
  unsigned short* vw1 = (unsigned short*)take((size_t)D * D * 2);
  unsigned short* vw2 = (unsigned short*)take((size_t)D * D * 2);
  unsigned short* qw1 = (unsigned short*)take((size_t)D * D * 2);
  unsigned short* qw2 = (unsigned short*)take((size_t)D * D * 2);
  unsigned short* mw1 = (unsigned short*)take((size_t)2 * D * 2 * D * 2);
  unsigned short* mw2 = (unsigned short*)take((size_t)D * 2 * D * 2);
  unsigned short* h_bf  = (unsigned short*)take((size_t)N * D * 2);
  unsigned short* hid   = (unsigned short*)take((size_t)N * 2 * D * 2);
  unsigned short* cat   = (unsigned short*)take((size_t)N * 2 * D * 2);
  unsigned short* k_bf  = (unsigned short*)take((size_t)N * D * 2);
  unsigned short* q_bf  = (unsigned short*)take((size_t)N * D * 2);
  unsigned short* v_bf  = (unsigned short*)take((size_t)N * D * 2);
  unsigned short* vt_bf = (unsigned short*)take((size_t)N * D * 2);
  float*          commf = (float*)take((size_t)N * D * 4);

  const int CT = 256;
  // weight + activation conversion (deterministic, cheap vs. GEMM work)
  cvt_f32_bf16<<<(D * D + CT - 1) / CT, CT, 0, stream>>>(f_kw1, kw1, D * D);
  cvt_f32_bf16<<<(D * D + CT - 1) / CT, CT, 0, stream>>>(f_kw2, kw2, D * D);
  cvt_f32_bf16<<<(D * D + CT - 1) / CT, CT, 0, stream>>>(f_vw1, vw1, D * D);
  cvt_f32_bf16<<<(D * D + CT - 1) / CT, CT, 0, stream>>>(f_vw2, vw2, D * D);
  cvt_f32_bf16<<<(D * D + CT - 1) / CT, CT, 0, stream>>>(f_qw1, qw1, D * D);
  cvt_f32_bf16<<<(D * D + CT - 1) / CT, CT, 0, stream>>>(f_qw2, qw2, D * D);
  cvt_f32_bf16<<<(4 * D * D + CT - 1) / CT, CT, 0, stream>>>(f_mw1, mw1, 4 * D * D);
  cvt_f32_bf16<<<(2 * D * D + CT - 1) / CT, CT, 0, stream>>>(f_mw2, mw2, 2 * D * D);
  cvt_f32_bf16<<<(N * D + CT - 1) / CT, CT, 0, stream>>>(hidden, h_bf, N * D);

  const dim3 gN512(4, N / 128);   // N=512 output GEMM grid
  const dim3 gN1024(8, N / 128);  // N=1024 output GEMM grid

  for (int hop = 0; hop < 3; ++hop) {
    if (hop > 0) {
      concat_comm_h<<<(N * 2 * D + CT - 1) / CT, CT, 0, stream>>>(commf, h_bf, cat, N * 2 * D);
      gemm_bias_act<<<gN1024, 256, 0, stream>>>(cat, mw1, msg_b1, hid, N, 2 * D, 2 * D, 1);
      gemm_bias_act<<<gN512, 256, 0, stream>>>(hid, mw2, msg_b2, h_bf, N, D, 2 * D, 0);
    }
    gemm_bias_act<<<gN512, 256, 0, stream>>>(h_bf, kw1, key_b1, hid, N, D, D, 1);
    gemm_bias_act<<<gN512, 256, 0, stream>>>(hid, kw2, key_b2, k_bf, N, D, D, 0);
    gemm_bias_act<<<gN512, 256, 0, stream>>>(h_bf, vw1, val_b1, hid, N, D, D, 1);
    gemm_bias_act<<<gN512, 256, 0, stream>>>(hid, vw2, val_b2, v_bf, N, D, D, 0);
    gemm_bias_act<<<gN512, 256, 0, stream>>>(h_bf, qw1, qry_b1, hid, N, D, D, 1);
    gemm_bias_act<<<gN512, 256, 0, stream>>>(hid, qw2, qry_b2, q_bf, N, D, D, 0);

    transpose_bf16<<<dim3(D / 32, N / 32), dim3(32, 8), 0, stream>>>(v_bf, vt_bf, N, D);

    float* outp = (hop == 2) ? (float*)d_out : commf;
    flash_attn<<<N / 64, 128, 0, stream>>>(q_bf, k_bf, vt_bf, outp, N);
  }
}
